// GATCL_14267881357872
// MI455X (gfx1250) — matrix operand order repaired and checked by reading, not compiled, not run on hardware
//
#include <hip/hip_runtime.h>
#include <hip/hip_bf16.h>
#include <stdint.h>

// ---------------------------------------------------------------------------
// GAT pipeline for MI455X (gfx1250, wave32, WMMA).
// Heavy ops (x@W, att@h x3, MLPs) use v_wmma_f32_16x16x32_bf16 with f32 acc.
// Attention is flash-style with a fixed per-row softmax shift derived from
// the monotonicity of the activation (no online rescale, no atomics).
// adj (256MB int32) is compressed once to a bitmask (8MB) via ballot.
// B-operand fragments stream global->LDS:
//   - prologue chunk via TDM tensor_load_to_lds (TENSORcnt + s_wait_tensorcnt)
//   - steady state via global_load_async_to_lds_b128 (ASYNCcnt double buffer)
// both overlapping with the VALU logit/exp work and the WMMA pipe.
// ---------------------------------------------------------------------------

typedef __bf16 bf16;
typedef __attribute__((ext_vector_type(16))) __bf16 v16bf;
typedef __attribute__((ext_vector_type(8)))  float  v8f;

#define N_NODES 8192
#define WORDS   256        // N_NODES / 32
#define IN_DIM  768
#define HT_TOT  512        // NHEAD * H1T fused columns of x@W
#define SEGS    2          // K-split segments for attention passes

__device__ __forceinline__ float lrelu_f(float x){ return x > 0.f ? x : 0.2f * x; }
__device__ __forceinline__ float elu_f(float x){ return x > 0.f ? x : __expf(x) - 1.f; }

// ---- WMMA fragment index maps (wave32, 16-bit operands; see CDNA5 ISA) ----
// A (16x32, M=lane&15): lanes 0-15 hold K=hv, lanes 16-31 hold K=16+hv
__device__ __forceinline__ int a_k(int lane, int hv){
  return hv + ((lane & 16) ? 16 : 0);
}
// B (32x16, N=lane&15): lanes 0-15 hold K=hv, lanes 16-31 hold K=16+hv
__device__ __forceinline__ int b_k(int lane, int hv){
  return hv + ((lane & 16) ? 16 : 0);
}
// C/D f32 (16x16): VGPR r -> row = r + (lane>=16 ? 8 : 0), col = lane&15

__device__ __forceinline__ v8f v8f_zero(){
  v8f z;
#pragma unroll
  for (int q = 0; q < 8; ++q) z[q] = 0.f;
  return z;
}

__device__ __forceinline__ v8f wmma_bf16(v16bf a, v16bf b, v8f c){
  return __builtin_amdgcn_wmma_f32_16x16x32_bf16(false, a, false, b, (short)0, c,
                                                 false, false);
}

// Async copy of one contiguous chunk (NLOAD x 512B) global -> LDS.
// LDS dest operand is the wave-relative LDS byte offset (HW adds LDS_BASE).
template<int NLOAD>
__device__ __forceinline__ void async_copy_chunk(const bf16* g, unsigned ldsbase,
                                                 int lane){
#pragma unroll
  for (int q = 0; q < NLOAD; ++q){
    unsigned off = (unsigned)(q * 512 + lane * 16);
    unsigned lo = ldsbase + off;
    unsigned long long ga = (unsigned long long)(uintptr_t)g + off;
    asm volatile("global_load_async_to_lds_b128 %0, %1, off"
                 :: "v"(lo), "v"(ga) : "memory");
  }
}

// TDM 1-D copy: len8 elements of 8 bytes, contiguous global -> LDS.
// D# per CDNA5 ISA:
//  group0: count=1 | lds_addr | global_addr[56:0] | type=2
//  group1: data_size=3 (8B), tensor_dim0=tile_dim0=len8, tensor_dim1=1,
//          no multicast / iterate / pad / atomic-barrier
//  trailing descriptor groups: zero (tile_dim3/4 = 0 => unused)
// This toolchain (clang-23 / therock-10.0) declares the 6-arg builtin:
//  (uint32x4, int32x8, int32x4, int32x4, int32x8, i32 cpol)
__device__ __forceinline__ void tdm_copy_1d(const void* g, unsigned lds_off,
                                            int len8){
  typedef unsigned int u32x4 __attribute__((ext_vector_type(4)));
  typedef int i32x8 __attribute__((ext_vector_type(8)));
  typedef int i32x4 __attribute__((ext_vector_type(4)));
  unsigned long long ga = (unsigned long long)(uintptr_t)g;
  u32x4 g0;
  g0[0] = 1u;                                           // count=1, load, user
  g0[1] = lds_off;                                      // lds_addr (bytes)
  g0[2] = (unsigned)ga;                                 // global_addr[31:0]
  g0[3] = (unsigned)((ga >> 32) & 0x01FFFFFFu) | (2u << 30); // addr[56:32]|type=2
  i32x8 g1;
  g1[0] = (int)(3u << 16);                              // data_size=3 (8 bytes)
  g1[1] = (int)(((unsigned)len8 & 0xFFFFu) << 16);      // tensor_dim0[15:0]
  g1[2] = (int)((((unsigned)len8 >> 16) & 0xFFFFu) | (1u << 16)); // td0 hi|td1=1
  g1[3] = (int)(((unsigned)len8 & 0xFFFFu) << 16);      // tile_dim0 = len8
  g1[4] = 0;                                            // tile_dim1/2 unused
  g1[5] = len8;                                         // dim0 stride (benign)
  g1[6] = 0;
  g1[7] = 0;
  i32x4 gz4;
  gz4[0] = 0; gz4[1] = 0; gz4[2] = 0; gz4[3] = 0;
  i32x8 gz8;
#pragma unroll
  for (int q = 0; q < 8; ++q) gz8[q] = 0;
  __builtin_amdgcn_tensor_load_to_lds(g0, g1, gz4, gz4, gz8, 0);
}

// ---------------------------------------------------------------------------
// adj (int32, N x N) -> bitmask (1 bit per edge) via wave32 ballot
// ---------------------------------------------------------------------------
__global__ void k_adjpack(const int* __restrict__ adj, unsigned* __restrict__ mask){
  size_t gid = (size_t)blockIdx.x * blockDim.x + threadIdx.x;
  unsigned b = __builtin_amdgcn_ballot_w32(adj[gid] > 0);
  if ((threadIdx.x & 31u) == 0u) mask[gid >> 5] = b;
}

// ---------------------------------------------------------------------------
// h = x @ Wcat  (8192x768 @ 768x512), written as bf16 B-fragments to hbuf.
// Block: 256 thr (8 waves), tile 32(M) x 256(N); wave -> 16x64 (4 C tiles).
// ---------------------------------------------------------------------------
__global__ void k_gemm_xw(const float* __restrict__ x,
                          const float* __restrict__ headW,   // (2,768,256)
                          bf16* __restrict__ hbuf){
  __shared__ __align__(32) bf16 As[2][32][16];
  __shared__ __align__(32) bf16 Bs[16][32][16];
  const int t = threadIdx.x, lane = t & 31, w = t >> 5;
  const int rowBase = blockIdx.x * 32;
  const int nBase   = blockIdx.y * 256;
  const int wm = w >> 2, wn = w & 3;

  v8f acc[4];
#pragma unroll
  for (int c = 0; c < 4; ++c) acc[c] = v8f_zero();

  for (int k0 = 0; k0 < IN_DIM; k0 += 32){
    if (t < 64){
      int tile = t >> 5, la = t & 31;
      int row = rowBase + tile * 16 + (la & 15);
#pragma unroll
      for (int hv = 0; hv < 16; ++hv)
        As[tile][la][hv] = (bf16)x[(size_t)row * IN_DIM + k0 + a_k(la, hv)];
    }
#pragma unroll
    for (int s = 0; s < 2; ++s){
      int slot = t * 2 + s;                 // 0..511
      int ft = slot >> 5, la = slot & 31;
      int nn = nBase + ft * 16 + (la & 15);
      int hd = nn >> 8, nl = nn & 255;
#pragma unroll
      for (int hv = 0; hv < 16; ++hv){
        int kk = k0 + b_k(la, hv);
        Bs[ft][la][hv] = (bf16)headW[(size_t)hd * IN_DIM * 256 + (size_t)kk * 256 + nl];
      }
    }
    __syncthreads();
    v16bf a = *(const v16bf*)&As[wm][lane][0];
#pragma unroll
    for (int c = 0; c < 4; ++c){
      v16bf b = *(const v16bf*)&Bs[wn * 4 + c][lane][0];
      acc[c] = wmma_bf16(a, b, acc[c]);
    }
    __syncthreads();
  }

  const int rrow = (lane & 16) ? 8 : 0, col = lane & 15;
#pragma unroll
  for (int c = 0; c < 4; ++c)
#pragma unroll
    for (int r = 0; r < 8; ++r){
      int i = rowBase + wm * 16 + r + rrow;           // node index (K dim later)
      int f = nBase + wn * 64 + c * 16 + col;         // feature (N dim later)
      int jc = i >> 5, ft = f >> 4;
      int lb = (f & 15) + (((i & 31) >= 16) ? 16 : 0);
      int hv = i & 15;
      hbuf[((size_t)jc * 32 + ft) * 512 + lb * 16 + hv] = (bf16)acc[c][r];
    }
}

// ---------------------------------------------------------------------------
// wh1/wh2 projections per head from the bf16 fragment buffer
// ---------------------------------------------------------------------------
__global__ void k_wh(const bf16* __restrict__ hbuf, const float* __restrict__ headA,
                     float* __restrict__ wh1h, float* __restrict__ wh2h){
  const int i = blockIdx.x, t = threadIdx.x;
  __shared__ float r1[256], r2[256];
  const int jc = i >> 5, hv = i & 15;
  const int lh = ((i & 31) >= 16) ? 16 : 0;
  for (int hd = 0; hd < 2; ++hd){
    int F = hd * 256 + t;
    int ft = F >> 4, lb = (F & 15) + lh;
    float v = (float)hbuf[((size_t)jc * 32 + ft) * 512 + lb * 16 + hv];
    r1[t] = v * headA[hd * 512 + t];
    r2[t] = v * headA[hd * 512 + 256 + t];
    __syncthreads();
    for (int off = 128; off > 0; off >>= 1){
      if (t < off){ r1[t] += r1[t + off]; r2[t] += r2[t + off]; }
      __syncthreads();
    }
    if (t == 0){ wh1h[hd * N_NODES + i] = r1[0]; wh2h[hd * N_NODES + i] = r2[0]; }
    __syncthreads();
  }
}

__global__ void k_max(const float* __restrict__ v, int n, float* __restrict__ outmax){
  __shared__ float red[256];
  int t = threadIdx.x;
  float m = -3.4e38f;
  for (int i = t; i < n; i += 256) m = fmaxf(m, v[i]);
  red[t] = m; __syncthreads();
  for (int off = 128; off > 0; off >>= 1){
    if (t < off) red[t] = fmaxf(red[t], red[t + off]);
    __syncthreads();
  }
  if (t == 0) outmax[0] = red[0];
}

// ---------------------------------------------------------------------------
// Flash-GAT: out_part = P @ h, l_part = row-sums of P, P computed on the fly.
// One wave per 16 rows; K (=node axis) split into SEGS deterministic slices.
// B fragments: prologue chunk via TDM, steady state via async double buffer.
// ACT: 0 = leaky_relu(0.2), 1 = elu.
// ---------------------------------------------------------------------------
template<int FEATS, int ACT>
__global__ void k_flash(const unsigned* __restrict__ mask,
                        const bf16* __restrict__ hfrag, int ftStride, int ftBase,
                        const float* __restrict__ wh1,
                        const float* __restrict__ wh2,
                        const float* __restrict__ wmax,
                        float* __restrict__ part,      // [SEGS][N][FEATS]
                        float* __restrict__ partl){    // [SEGS][N]
  constexpr int NT = FEATS / 16;
  constexpr int NLOAD = (NT * 1024) / 512;   // async b128 instructions / chunk
  __shared__ __align__(32) bf16 Bl[2][NT * 512];

  const int lane = threadIdx.x;
  const int row0 = blockIdx.x * 16;
  const int seg  = blockIdx.y;
  const int myrow = row0 + (lane & 15);

  const float mx   = wmax[0];
  const float wh1r = wh1[myrow];
  const float shift = (ACT == 0) ? lrelu_f(wh1r + mx) : elu_f(wh1r + mx);

  const unsigned ldsB0 = (unsigned)(uintptr_t)&Bl[0][0];
  const unsigned ldsB1 = (unsigned)(uintptr_t)&Bl[1][0];

  v8f acc[NT];
#pragma unroll
  for (int tnt = 0; tnt < NT; ++tnt) acc[tnt] = v8f_zero();
  float lsum = 0.f;

  const int jPerSeg = N_NODES / SEGS;
  const int j0s = seg * jPerSeg, j0e = j0s + jPerSeg;

  // kick off the first chunk's B-fragment copy with the Tensor Data Mover
  tdm_copy_1d(hfrag + ((size_t)(j0s >> 5) * ftStride + ftBase) * 512,
              ((j0s >> 5) & 1) ? ldsB1 : ldsB0, NT * 128);

  for (int j0 = j0s; j0 < j0e; j0 += 32){
    const int parity = (j0 >> 5) & 1;
    const bool more = (j0 + 32) < j0e;
    if (more)
      async_copy_chunk<NLOAD>(
          hfrag + ((size_t)((j0 + 32) >> 5) * ftStride + ftBase) * 512,
          parity ? ldsB0 : ldsB1, lane);

    // compute P tile while the async/TDM engines stream B
    if (j0 + 256 < j0e)
      __builtin_prefetch(&mask[(size_t)myrow * WORDS + ((j0 + 256) >> 5)], 0, 3);
    unsigned mword = mask[(size_t)myrow * WORDS + (j0 >> 5)];
    v16bf a;
#pragma unroll
    for (int hv = 0; hv < 16; ++hv){
      int k = a_k(lane, hv);
      float s = wh1r + wh2[j0 + k];
      s = (ACT == 0) ? lrelu_f(s) : elu_f(s);
      float p = ((mword >> k) & 1u) ? __expf(s - shift) : 0.f;
      lsum += p;
      a[hv] = (bf16)p;
    }

    // older async batch done when <= NLOAD remain (in-order completion)
    if (more) asm volatile("s_wait_asynccnt %0" :: "n"(NLOAD) : "memory");
    else      asm volatile("s_wait_asynccnt 0" ::: "memory");
    if (j0 == j0s){                       // prologue chunk came via TDM
      __builtin_amdgcn_s_wait_tensorcnt(0);
      asm volatile("" ::: "memory");
    }

    const bf16* cur = &Bl[parity][0];
#pragma unroll
    for (int tnt = 0; tnt < NT; ++tnt){
      v16bf b = *(const v16bf*)(cur + tnt * 512 + lane * 16);
      acc[tnt] = wmma_bf16(a, b, acc[tnt]);
    }
  }

  lsum += __shfl_xor(lsum, 16, 32);
  if (lane < 16) partl[(size_t)seg * N_NODES + myrow] = lsum;

  float* po = part + ((size_t)seg * N_NODES + row0) * FEATS;
  const int rrow = (lane & 16) ? 8 : 0, col = lane & 15;
#pragma unroll
  for (int tnt = 0; tnt < NT; ++tnt)
#pragma unroll
    for (int r = 0; r < 8; ++r)
      po[(size_t)(r + rrow) * FEATS + tnt * 16 + col] = acc[tnt][r];
}

// ---------------------------------------------------------------------------
// Combine K-segments, normalize softmax, elu, l2norm, +bias
// ---------------------------------------------------------------------------
__global__ void k_combine(const float* __restrict__ part, const float* __restrict__ partl,
                          const float* __restrict__ bias, float* __restrict__ outp,
                          int feats){
  const int row = blockIdx.x, t = threadIdx.x;
  __shared__ float red[256];
  float l = 0.f;
  for (int s = 0; s < SEGS; ++s) l += partl[(size_t)s * N_NODES + row];
  float v = 0.f;
  if (t < feats){
    for (int s = 0; s < SEGS; ++s) v += part[((size_t)s * N_NODES + row) * feats + t];
    v = elu_f(v / l);
  }
  red[t] = v * v; __syncthreads();
  for (int off = 128; off > 0; off >>= 1){
    if (t < off) red[t] += red[t + off];
    __syncthreads();
  }
  float sc = 1.f / fmaxf(sqrtf(red[0]), 1e-12f);
  if (t < feats) outp[(size_t)row * feats + t] = v * sc + bias[t];
}

// ---------------------------------------------------------------------------
// z = elu(mean(heads)), 1D VALID conv (K=129) -> h2 (128), conv wh dots,
// write h2 as bf16 B-fragments for the second attention pass.
// ---------------------------------------------------------------------------
__global__ void k_mid(const float* __restrict__ headout, const float* __restrict__ convw,
                      const float* __restrict__ convcb, const float* __restrict__ conva,
                      bf16* __restrict__ hbuf2, float* __restrict__ whc1,
                      float* __restrict__ whc2){
  const int i = blockIdx.x, t = threadIdx.x;
  __shared__ float z[256], h2s[128], r1[128], r2[128];
  float m = 0.5f * (headout[(size_t)i * 256 + t] +
                    headout[(size_t)(N_NODES + i) * 256 + t]);
  z[t] = elu_f(m);
  __syncthreads();
  if (t < 128){
    float s = convcb[0];
    for (int k = 0; k < 129; ++k) s += z[t + k] * convw[k];
    h2s[t] = s;
    r1[t] = s * conva[t];
    r2[t] = s * conva[128 + t];
  }
  __syncthreads();
  for (int off = 64; off > 0; off >>= 1){
    if (t < off){ r1[t] += r1[t + off]; r2[t] += r2[t + off]; }
    __syncthreads();
  }
  if (t == 0){ whc1[i] = r1[0]; whc2[i] = r2[0]; }
  if (t < 128){
    int jc = i >> 5, ft = t >> 4;
    int lb = (t & 15) + (((i & 31) >= 16) ? 16 : 0);
    int hv = i & 15;
    hbuf2[((size_t)jc * 8 + ft) * 512 + lb * 16 + hv] = (bf16)h2s[t];
  }
}

// ---------------------------------------------------------------------------
// Fused 2-layer MLP: out = elu(elu(in@w1+b1)@w2+b2). Block: 128 rows, 8 waves.
// Intermediate re-fragmented through wave-private LDS (C-layout -> A-layout).
// ---------------------------------------------------------------------------
__global__ void k_mlp(const float* __restrict__ inp, const float* __restrict__ w1,
                      const float* __restrict__ b1, const float* __restrict__ w2,
                      const float* __restrict__ b2, float* __restrict__ outp){
  __shared__ __align__(32) bf16 W1s[4][8][32][16];
  __shared__ __align__(32) bf16 W2s[4][4][32][16];
  __shared__ __align__(32) bf16 Ablk[8][4][32][16];
  const int t = threadIdx.x, lane = t & 31, w = t >> 5;

  for (int slot = t; slot < 4 * 8 * 32; slot += 256){
    int kc = slot >> 8, ft = (slot >> 5) & 7, la = slot & 31;
#pragma unroll
    for (int hv = 0; hv < 16; ++hv){
      int k = kc * 32 + b_k(la, hv), n = ft * 16 + (la & 15);
      W1s[kc][ft][la][hv] = (bf16)w1[k * 128 + n];
    }
  }
  for (int slot = t; slot < 4 * 4 * 32; slot += 256){
    int kc = slot >> 7, ft = (slot >> 5) & 3, la = slot & 31;
#pragma unroll
    for (int hv = 0; hv < 16; ++hv){
      int k = kc * 32 + b_k(la, hv), n = ft * 16 + (la & 15);
      W2s[kc][ft][la][hv] = (bf16)w2[k * 64 + n];
    }
  }
  __syncthreads();

  const int r0 = blockIdx.x * 128 + w * 16;
  v8f acc1[8];
#pragma unroll
  for (int ft = 0; ft < 8; ++ft) acc1[ft] = v8f_zero();
  for (int kc = 0; kc < 4; ++kc){
    v16bf a;
    int row = r0 + (lane & 15);
#pragma unroll
    for (int hv = 0; hv < 16; ++hv)
      a[hv] = (bf16)inp[(size_t)row * 128 + kc * 32 + a_k(lane, hv)];
#pragma unroll
    for (int ft = 0; ft < 8; ++ft){
      v16bf b = *(const v16bf*)&W1s[kc][ft][lane][0];
      acc1[ft] = wmma_bf16(a, b, acc1[ft]);
    }
  }

  const int rrow = (lane & 16) ? 8 : 0, col = lane & 15;
#pragma unroll
  for (int ft = 0; ft < 8; ++ft)
#pragma unroll
    for (int r = 0; r < 8; ++r){
      float u = elu_f(acc1[ft][r] + b1[ft * 16 + col]);
      int F = ft * 16 + col, lr = r + rrow;
      int kc2 = F >> 5, k32 = F & 31;
      int lh, hv;
      lh = (k32 >= 16) ? 16 : 0; hv = k32 & 15;
      Ablk[w][kc2][lr + lh][hv] = (bf16)u;   // wave-private; DS in-order
    }

  v8f acc2[4];
#pragma unroll
  for (int ft = 0; ft < 4; ++ft) acc2[ft] = v8f_zero();
  for (int kc = 0; kc < 4; ++kc){
    v16bf a2 = *(const v16bf*)&Ablk[w][kc][lane][0];
#pragma unroll
    for (int ft = 0; ft < 4; ++ft){
      v16bf b = *(const v16bf*)&W2s[kc][ft][lane][0];
      acc2[ft] = wmma_bf16(a2, b, acc2[ft]);
    }
  }
#pragma unroll
  for (int ft = 0; ft < 4; ++ft)
#pragma unroll
    for (int r = 0; r < 8; ++r){
      float v = elu_f(acc2[ft][r] + b2[ft * 16 + col]);
      outp[(size_t)(r0 + r + rrow) * 64 + ft * 16 + col] = v;
    }
}

// ---------------------------------------------------------------------------
// pred[e] = dot(tf[i], tg[j])
// ---------------------------------------------------------------------------
__global__ void k_pred(const int* __restrict__ ts, const float* __restrict__ tf,
                       const float* __restrict__ tg, float* __restrict__ outp, int E){
  int e = blockIdx.x * blockDim.x + threadIdx.x;
  if (e >= E) return;
  int i = ts[e * 2], j = ts[e * 2 + 1];
  const float4* pa = (const float4*)(tf + (size_t)i * 64);
  const float4* pb = (const float4*)(tg + (size_t)j * 64);
  float s = 0.f;
#pragma unroll
  for (int q = 0; q < 16; ++q){
    float4 a = pa[q], b = pb[q];
    s += a.x * b.x + a.y * b.y + a.z * b.z + a.w * b.w;
  }
  outp[e] = s;
}

// ---------------------------------------------------------------------------
extern "C" void kernel_launch(void* const* d_in, const int* in_sizes, int n_in,
                              void* d_out, int out_size, void* d_ws, size_t ws_size,
                              hipStream_t stream){
  const float* x      = (const float*)d_in[0];
  const int*   adj    = (const int*)d_in[1];
  const int*   ts     = (const int*)d_in[2];
  const float* headW  = (const float*)d_in[3];
  const float* headA  = (const float*)d_in[4];
  const float* headB  = (const float*)d_in[5];
  const float* convw  = (const float*)d_in[6];
  const float* convcb = (const float*)d_in[7];
  const float* conva  = (const float*)d_in[8];
  const float* convb  = (const float*)d_in[9];
  const float* tfw1   = (const float*)d_in[10];
  const float* tfb1   = (const float*)d_in[11];
  const float* tfw2   = (const float*)d_in[12];
  const float* tfb2   = (const float*)d_in[13];
  const float* tgw1   = (const float*)d_in[14];
  const float* tgb1   = (const float*)d_in[15];
  const float* tgw2   = (const float*)d_in[16];
  const float* tgb2   = (const float*)d_in[17];
  float* outp = (float*)d_out;
  const int E = in_sizes[2] / 2;

  // workspace carve-up (256B aligned)
  char* p = (char*)d_ws;
  auto carve = [&](size_t bytes) -> char* {
    char* r = p; p += (bytes + 255) & ~(size_t)255; return r;
  };
  unsigned* mask  = (unsigned*)carve((size_t)N_NODES * WORDS * 4);          // 8 MB
  bf16* hbuf      = (bf16*)carve((size_t)N_NODES * HT_TOT * 2);             // 8 MB
  bf16* hbuf2     = (bf16*)carve((size_t)N_NODES * 128 * 2);                // 2 MB
  float* wh1h     = (float*)carve((size_t)2 * N_NODES * 4);
  float* wh2h     = (float*)carve((size_t)2 * N_NODES * 4);
  float* whc1     = (float*)carve((size_t)N_NODES * 4);
  float* whc2     = (float*)carve((size_t)N_NODES * 4);
  float* wmax     = (float*)carve(4 * 4);
  float* headout  = (float*)carve((size_t)2 * N_NODES * 256 * 4);           // 16 MB
  float* embed    = (float*)carve((size_t)N_NODES * 128 * 4);               // 4 MB
  float* tfbuf    = (float*)carve((size_t)N_NODES * 64 * 4);                // 2 MB
  float* tgbuf    = (float*)carve((size_t)N_NODES * 64 * 4);                // 2 MB
  float* part     = (float*)carve((size_t)SEGS * N_NODES * 256 * 4);        // 16 MB
  float* partl    = (float*)carve((size_t)SEGS * N_NODES * 4);

  // 1) adjacency -> bitmask
  k_adjpack<<<dim3((N_NODES * (size_t)N_NODES) / 256), 256, 0, stream>>>(adj, mask);

  // 2) h = x @ Wcat (bf16 B-fragments)
  k_gemm_xw<<<dim3(N_NODES / 32, HT_TOT / 256), 256, 0, stream>>>(x, headW, hbuf);

  // 3) logit projections + global max of wh2 per head
  k_wh<<<dim3(N_NODES), 256, 0, stream>>>(hbuf, headA, wh1h, wh2h);
  k_max<<<dim3(1), 256, 0, stream>>>(wh2h, N_NODES, wmax + 0);
  k_max<<<dim3(1), 256, 0, stream>>>(wh2h + N_NODES, N_NODES, wmax + 1);

  // 4) per-head flash attention + epilogue
  for (int hd = 0; hd < 2; ++hd){
    k_flash<256, 0><<<dim3(N_NODES / 16, SEGS), 32, 0, stream>>>(
        mask, hbuf, 32, hd * 16, wh1h + hd * N_NODES, wh2h + hd * N_NODES,
        wmax + hd, part, partl);
    k_combine<<<dim3(N_NODES), 256, 0, stream>>>(
        part, partl, headB + hd * 256, headout + (size_t)hd * N_NODES * 256, 256);
  }

  // 5) mean/elu/conv + conv-GAT projections
  k_mid<<<dim3(N_NODES), 256, 0, stream>>>(headout, convw, convcb, conva,
                                           hbuf2, whc1, whc2);
  k_max<<<dim3(1), 256, 0, stream>>>(whc2, N_NODES, wmax + 2);

  // 6) conv-GAT flash attention (elu logits) + epilogue -> embed
  k_flash<128, 1><<<dim3(N_NODES / 16, SEGS), 32, 0, stream>>>(
      mask, hbuf2, 8, 0, whc1, whc2, wmax + 2, part, partl);
  k_combine<<<dim3(N_NODES), 256, 0, stream>>>(part, partl, convb, embed, 128);

  // 7) twin MLPs
  k_mlp<<<dim3(N_NODES / 128), 256, 0, stream>>>(embed, tfw1, tfb1, tfw2, tfb2, tfbuf);
  k_mlp<<<dim3(N_NODES / 128), 256, 0, stream>>>(embed, tgw1, tgb1, tgw2, tgb2, tgbuf);

  // 8) edge predictions
  k_pred<<<dim3((E + 255) / 256), 256, 0, stream>>>(ts, tfbuf, tgbuf, outp, E);
}
